// RRN_20005957665474
// MI455X (gfx1250) — compile-verified, matrix operand order checked by reading, not run on hardware
//
#include <hip/hip_runtime.h>
#include <math.h>

// ---------------- problem constants (from reference) ----------------
#define NN 100000   // individuals
#define DD 128      // embedding dim
#define KK 64       // classes
#define LL 8        // relation groups
#define EE 50000    // edges per group
#define DKC (DD + KK)   // 192

// rel kernel tiling: 512 threads, stage 8 weight matrices (256KB) in LDS,
// loop over REL_TILES tiles of 16 edges each with double-buffered staging.
#define REL_TILES 25                     // 400 edges per block
#define REL_BPL   (EE / 16 / REL_TILES)  // 125 blocks per layer
#define CLS_TILES 5                      // 80 rows per block
#define CLS_GRID  (NN / (16 * CLS_TILES))// 1250

// ---------------- CDNA5 WMMA types ----------------
typedef __attribute__((ext_vector_type(16))) __bf16 v16bf;
typedef __attribute__((ext_vector_type(8)))  __bf16 v8bf;
typedef __attribute__((ext_vector_type(4)))  __bf16 v4bf;
typedef __attribute__((ext_vector_type(8)))  float  v8f;

__device__ __forceinline__ v8f wmma_bf16(v16bf a, v16bf b, v8f c) {
    // D = A(16x32 bf16) * B(32x16 bf16) + C(16x16 f32)
    return __builtin_amdgcn_wmma_f32_16x16x32_bf16(
        /*neg_a=*/false, a, /*neg_b=*/false, b,
        /*c_mod=*/(short)0, c, /*reuse_a=*/false, /*reuse_b=*/false);
}

// Lane fragment for a 32-wide K chunk (ISA 7.12.2 16-bit layout):
// lanes 0-15 hold K {k..k+7, k+16..k+23}; lanes 16-31 hold {k+8..k+15, k+24..k+31}.
// Caller passes p = row + k + lane_hi*8; we read p[0..7] and p[16..23].
__device__ __forceinline__ v16bf frag16(const __bf16* p) {
    v8bf lo = *(const v8bf*)(p);
    v8bf hi = *(const v8bf*)(p + 16);
    v16bf r;
#pragma unroll
    for (int i = 0; i < 8; ++i) { r[i] = lo[i]; r[i + 8] = hi[i]; }
    return r;
}

__device__ __forceinline__ float sigmoidf_(float x) {
    return 1.0f / (1.0f + expf(-x));
}

// ---------------- f32 -> bf16 conversion ----------------
__global__ void cvt_bf16(const float* __restrict__ s, __bf16* __restrict__ d, int n) {
    int i = blockIdx.x * 256 + threadIdx.x;
    if (i < n) d[i] = (__bf16)s[i];
}

// ---------------- ClassUpdate ------------------------------------------------
// upd = l2norm(emb + sigmoid(cat@V^T) * relu(cat@W^T)),  cat = [emb | memb].
// Both class matrices staged in LDS once; block loops CLS_TILES tiles of 16 rows.
__global__ __launch_bounds__(256) void class_kernel(
    const float* __restrict__ emb, const float* __restrict__ memb,
    const __bf16* __restrict__ cV, const __bf16* __restrict__ cW,
    float* __restrict__ upd, __bf16* __restrict__ updh, float* __restrict__ acc)
{
    extern __shared__ char smem[];
    __bf16* cVl  = (__bf16*)smem;            // 128*192
    __bf16* cWl  = cVl + DD * DKC;           // 128*192
    __bf16* cat  = cWl + DD * DKC;           // 16*192
    float*  tmp  = (float*)(cat + 16 * DKC); // 16*128
    float*  norms = tmp + 16 * DD;           // 16

    const int tid = threadIdx.x;

    // stage class weights (bf16) into LDS
    for (int j = tid * 8; j < DD * DKC; j += 256 * 8) {
        *(v8bf*)(cVl + j) = *(const v8bf*)(cV + j);
        *(v8bf*)(cWl + j) = *(const v8bf*)(cW + j);
    }
    __syncthreads();

    const int wave = tid >> 5, lane = tid & 31;
    const int lane_hi = lane >> 4, nl = lane & 15;
    const int nbase = wave * 16;

    for (int t = 0; t < CLS_TILES; ++t) {
        const int row0 = (blockIdx.x * CLS_TILES + t) * 16;
        __syncthreads();   // protect cat/tmp from previous tile's readers

        // stage cat = [emb | memberships] as bf16 (12 elems per thread)
        {
            int r = tid >> 4, c0 = (tid & 15) * 12;
            int row = row0 + r;
#pragma unroll
            for (int i = 0; i < 12; ++i) {
                int c = c0 + i;
                float v = (c < DD) ? emb[(size_t)row * DD + c]
                                   : memb[(size_t)row * KK + (c - DD)];
                cat[r * DKC + c] = (__bf16)v;
            }
        }
        __syncthreads();

        v8f aV = {}; v8f aW = {};
#pragma unroll
        for (int k = 0; k < DKC; k += 32) {
            int ko = k + lane_hi * 8;
            v16bf a  = frag16(cat + nl * DKC + ko);               // A row m = nl
            v16bf bv = frag16(cVl + (nbase + nl) * DKC + ko);     // B col f = nbase+nl
            v16bf bw = frag16(cWl + (nbase + nl) * DKC + ko);
            aV = wmma_bf16(a, bv, aV);
            aW = wmma_bf16(a, bw, aW);
        }

        // epilogue: emb + gate*dir  (C layout: reg r -> M = r + 8*lane_hi, N = nl)
#pragma unroll
        for (int r = 0; r < 8; ++r) {
            int m = r + lane_hi * 8;
            int f = nbase + nl;
            float gate = sigmoidf_(aV[r]);
            float dir  = aW[r] > 0.0f ? aW[r] : 0.0f;
            tmp[m * DD + f] = emb[(size_t)(row0 + m) * DD + f] + gate * dir;
        }
        __syncthreads();

        if (tid < 16) {
            float s = 0.0f;
            for (int c = 0; c < DD; ++c) { float v = tmp[tid * DD + c]; s += v * v; }
            float nrm = sqrtf(s);
            norms[tid] = nrm > 1e-12f ? nrm : 1e-12f;
        }
        __syncthreads();

        {
            int r = tid >> 4, c0 = (tid & 15) * 8;
            float inv = 1.0f / norms[r];
            size_t base = (size_t)(row0 + r) * DD + c0;
#pragma unroll
            for (int i = 0; i < 8; ++i) {
                float u = tmp[r * DD + c0 + i] * inv;
                upd[base + i]  = u;
                updh[base + i] = (__bf16)u;
                acc[base + i]  = 0.0f;   // reset scatter accumulator
            }
        }
    }
}

// ---------------- RelationUpdates (sub & obj fused) -------------------------
// 512 threads / 16 waves.  Block stages all 8 relation matrices of its layer
// into 256KB LDS once, then loops over REL_TILES tiles of 16 edges with
// double-buffered edge staging: gathers for tile t+1 and gate-scalar dots for
// tile t run concurrently with tile t's 16-WMMA chain.
// Wave w: module = w>>3 (0=sub,1=obj), f-tile = w&7.
__global__ __launch_bounds__(512) void rel_kernel(
    const __bf16* __restrict__ updh,
    const __bf16* __restrict__ sVs, const __bf16* __restrict__ sVo,
    const __bf16* __restrict__ sWs, const __bf16* __restrict__ sWo,
    const __bf16* __restrict__ oVs, const __bf16* __restrict__ oVo,
    const __bf16* __restrict__ oWs, const __bf16* __restrict__ oWo,
    const float* __restrict__ sub_w, const float* __restrict__ obj_w,
    const int* __restrict__ s_idx, const int* __restrict__ o_idx,
    float* __restrict__ acc)
{
    extern __shared__ char smem[];
    __bf16* Wl    = (__bf16*)smem;             // 8 * 128*128 (sVs,sVo,sWs,sWo,oVs,oVo,oWs,oWo)
    __bf16* esb   = Wl + 8 * DD * DD;          // 2 * 16*128 (double-buffered)
    __bf16* eob   = esb + 2 * 16 * DD;         // 2 * 16*128
    int*    srowb = (int*)(eob + 2 * 16 * DD); // 2 * 16
    int*    orowb = srowb + 32;                // 2 * 16
    float*  gpair = (float*)(orowb + 32);      // 2 * 32  ([buf][0..15]=g_sub, [16..31]=g_obj)

    const int tid  = threadIdx.x;
    const int l    = blockIdx.x / REL_BPL;
    const int blk  = blockIdx.x % REL_BPL;
    const size_t lE = (size_t)l * EE;
    const int tile0 = blk * REL_TILES;

    // ---- stage 8 weight matrices for layer l into LDS (256KB) ----
    {
        const __bf16* src[8] = { sVs, sVo, sWs, sWo, oVs, oVo, oWs, oWo };
#pragma unroll
        for (int i = 0; i < 8; ++i) {
            const __bf16* s = src[i] + (size_t)l * DD * DD;
            __bf16* d = Wl + i * DD * DD;
            for (int j = tid * 8; j < DD * DD; j += 512 * 8)
                *(v8bf*)(d + j) = *(const v8bf*)(s + j);
        }
    }

    const int wave = tid >> 5, lane = tid & 31;
    const int module = wave >> 3;            // 0 = sub, 1 = obj
    const int nbase  = (wave & 7) * 16;
    const int lane_hi = lane >> 4, nl = lane & 15;

    const __bf16* bVs = Wl + (module * 4 + 0) * DD * DD + (nbase + nl) * DD;
    const __bf16* bVo = Wl + (module * 4 + 1) * DD * DD + (nbase + nl) * DD;
    const __bf16* bWs = Wl + (module * 4 + 2) * DD * DD + (nbase + nl) * DD;
    const __bf16* bWo = Wl + (module * 4 + 3) * DD * DD + (nbase + nl) * DD;

    // gather one 16-edge tile into buffer `buf` and zero its gate scalars
    auto gather = [&](int t, int buf) {
        int r = tid >> 5, c0 = (tid & 31) * 4;          // 4 bf16 (8B) per thread
        int e0 = (tile0 + t) * 16;
        int si = s_idx[lE + e0 + r];
        int oi = o_idx[lE + e0 + r];
        if ((tid & 31) == 0) { srowb[buf * 16 + r] = si; orowb[buf * 16 + r] = oi; }
        *(v4bf*)(esb + buf * 16 * DD + r * DD + c0) = *(const v4bf*)(updh + (size_t)si * DD + c0);
        *(v4bf*)(eob + buf * 16 * DD + r * DD + c0) = *(const v4bf*)(updh + (size_t)oi * DD + c0);
        if (tid < 32) gpair[buf * 32 + tid] = 0.0f;
    };

    // partial dots g = e_o . w (sub for tid<256, obj for tid>=256) into gpair[buf]
    auto dots = [&](int buf) {
        int t2 = tid & 255;
        int r2 = t2 >> 4, d0 = (t2 & 15) * 8;
        const float* w = (tid < 256 ? sub_w : obj_w) + (size_t)l * DD;
        const __bf16* eor = eob + buf * 16 * DD + r2 * DD + d0;
        float s = 0.0f;
#pragma unroll
        for (int i = 0; i < 8; ++i) s += (float)eor[i] * w[d0 + i];
        atomicAdd(gpair + buf * 32 + (tid < 256 ? 0 : 16) + r2, s);   // ds_add_f32
    };

    gather(0, 0);
    __syncthreads();   // weights + tile-0 staging visible

    for (int t = 0; t < REL_TILES; ++t) {
        const int cur = t & 1, nxt = cur ^ 1;

        // ---- phase A: dots(t) + gather(t+1) + WMMA(t), no internal barrier ----
        dots(cur);
        if (t + 1 < REL_TILES) gather(t + 1, nxt);
        if (t + 2 < REL_TILES && tid == 0) {
            __builtin_prefetch((const void*)(s_idx + lE + (size_t)(tile0 + t + 2) * 16), 0, 1);
            __builtin_prefetch((const void*)(o_idx + lE + (size_t)(tile0 + t + 2) * 16), 0, 1);
        }

        const __bf16* esT = esb + cur * 16 * DD;
        const __bf16* eoT = eob + cur * 16 * DD;
        v8f aG = {}, aD = {};
#pragma unroll
        for (int k = 0; k < DD; k += 32) {
            int ko = k + lane_hi * 8;
            v16bf As = frag16(esT + nl * DD + ko);
            v16bf Ao = frag16(eoT + nl * DD + ko);
            aG = wmma_bf16(As, frag16(bVs + ko), aG);
            aG = wmma_bf16(Ao, frag16(bVo + ko), aG);
            aD = wmma_bf16(As, frag16(bWs + ko), aD);
            aD = wmma_bf16(Ao, frag16(bWo + ko), aD);
        }
        __syncthreads();   // gpair[cur] complete; tile t+1 staged

        // ---- epilogue + scatter: u = sigmoid(gate) * relu(dir + g * e_s) ----
#pragma unroll
        for (int r = 0; r < 8; ++r) {
            int m = r + lane_hi * 8;
            int f = nbase + nl;
            float esv = (float)esT[m * DD + f];
            float g   = gpair[cur * 32 + module * 16 + m];
            float d   = aD[r] + g * esv; d = d > 0.0f ? d : 0.0f;
            float u   = d * sigmoidf_(aG[r]);
            int row   = module ? orowb[cur * 16 + m] : srowb[cur * 16 + m];
            __hip_atomic_fetch_add(&acc[(size_t)row * DD + f], u,
                                   __ATOMIC_RELAXED, __HIP_MEMORY_SCOPE_AGENT);
        }
        __syncthreads();   // epilogue readers done before gather overwrites `cur`
    }
}

// ---------------- finalize: out = l2norm(upd + acc) -------------------------
__global__ __launch_bounds__(256) void fin_kernel(
    const float* __restrict__ upd, const float* __restrict__ acc,
    float* __restrict__ out)
{
    const int wave = threadIdx.x >> 5, lane = threadIdx.x & 31;
    const int row  = blockIdx.x * 8 + wave;
    const size_t base = (size_t)row * DD;

    float v[4]; float s = 0.0f;
#pragma unroll
    for (int i = 0; i < 4; ++i) {
        int c = lane + i * 32;
        v[i] = upd[base + c] + acc[base + c];
        s += v[i] * v[i];
    }
#pragma unroll
    for (int off = 16; off > 0; off >>= 1) s += __shfl_xor(s, off, 32);
    float nrm = sqrtf(s); nrm = nrm > 1e-12f ? nrm : 1e-12f;
    float inv = 1.0f / nrm;
#pragma unroll
    for (int i = 0; i < 4; ++i) out[base + lane + i * 32] = v[i] * inv;
}

// ---------------- host orchestration ----------------------------------------
extern "C" void kernel_launch(void* const* d_in, const int* in_sizes, int n_in,
                              void* d_out, int out_size, void* d_ws, size_t ws_size,
                              hipStream_t stream) {
    (void)in_sizes; (void)n_in; (void)out_size; (void)ws_size;

    const float* embedding_m = (const float*)d_in[0];
    const float* memberships = (const float*)d_in[1];
    const float* class_V = (const float*)d_in[2];
    const float* class_W = (const float*)d_in[3];
    const float* sub_Vs = (const float*)d_in[4];
    const float* sub_Vo = (const float*)d_in[5];
    const float* sub_Ws = (const float*)d_in[6];
    const float* sub_Wo = (const float*)d_in[7];
    const float* sub_w  = (const float*)d_in[8];
    const float* obj_Vs = (const float*)d_in[9];
    const float* obj_Vo = (const float*)d_in[10];
    const float* obj_Ws = (const float*)d_in[11];
    const float* obj_Wo = (const float*)d_in[12];
    const float* obj_w  = (const float*)d_in[13];
    const int*   s_idx  = (const int*)d_in[14];
    const int*   o_idx  = (const int*)d_in[15];
    float* out = (float*)d_out;

    // workspace carve-out (256B aligned slices)
    char* ws = (char*)d_ws;
    size_t off = 0;
    auto carve = [&](size_t bytes) {
        void* p = ws + off;
        off = (off + bytes + 255) & ~(size_t)255;
        return p;
    };
    float*  emb_ws = (float*)carve((size_t)NN * DD * 4);
    float*  upd    = (float*)carve((size_t)NN * DD * 4);
    float*  acc    = (float*)carve((size_t)NN * DD * 4);
    __bf16* updh   = (__bf16*)carve((size_t)NN * DD * 2);
    const size_t relsz = (size_t)LL * DD * DD;
    __bf16* h_sVs = (__bf16*)carve(relsz * 2);
    __bf16* h_sVo = (__bf16*)carve(relsz * 2);
    __bf16* h_sWs = (__bf16*)carve(relsz * 2);
    __bf16* h_sWo = (__bf16*)carve(relsz * 2);
    __bf16* h_oVs = (__bf16*)carve(relsz * 2);
    __bf16* h_oVo = (__bf16*)carve(relsz * 2);
    __bf16* h_oWs = (__bf16*)carve(relsz * 2);
    __bf16* h_oWo = (__bf16*)carve(relsz * 2);
    const size_t clssz = (size_t)DD * DKC;
    __bf16* h_cV = (__bf16*)carve(clssz * 2);
    __bf16* h_cW = (__bf16*)carve(clssz * 2);

    // weight conversions (deterministic & idempotent)
    const int cvb = 256;
    int nrel = (int)relsz;
    cvt_bf16<<<(nrel + cvb - 1) / cvb, cvb, 0, stream>>>(sub_Vs, h_sVs, nrel);
    cvt_bf16<<<(nrel + cvb - 1) / cvb, cvb, 0, stream>>>(sub_Vo, h_sVo, nrel);
    cvt_bf16<<<(nrel + cvb - 1) / cvb, cvb, 0, stream>>>(sub_Ws, h_sWs, nrel);
    cvt_bf16<<<(nrel + cvb - 1) / cvb, cvb, 0, stream>>>(sub_Wo, h_sWo, nrel);
    cvt_bf16<<<(nrel + cvb - 1) / cvb, cvb, 0, stream>>>(obj_Vs, h_oVs, nrel);
    cvt_bf16<<<(nrel + cvb - 1) / cvb, cvb, 0, stream>>>(obj_Vo, h_oVo, nrel);
    cvt_bf16<<<(nrel + cvb - 1) / cvb, cvb, 0, stream>>>(obj_Ws, h_oWs, nrel);
    cvt_bf16<<<(nrel + cvb - 1) / cvb, cvb, 0, stream>>>(obj_Wo, h_oWo, nrel);
    int ncls = (int)clssz;
    cvt_bf16<<<(ncls + cvb - 1) / cvb, cvb, 0, stream>>>(class_V, h_cV, ncls);
    cvt_bf16<<<(ncls + cvb - 1) / cvb, cvb, 0, stream>>>(class_W, h_cW, ncls);

    // dynamic LDS sizes
    const size_t cls_lds = (size_t)2 * DD * DKC * 2 + 16 * DKC * 2 + 16 * DD * 4 + 16 * 4;   // 112704
    const size_t rel_lds = (size_t)8 * DD * DD * 2            // weights        262144
                         + (size_t)2 * 2 * 16 * DD * 2        // es/eo x2 bufs   16384
                         + 2 * 16 * 4 * 2                     // srow/orow x2      256
                         + 2 * 32 * 4;                        // gate scalars x2   256
                                                              // total          279040

    const int rel_grid = LL * REL_BPL;   // 1000
    const int fin_grid = NN / 8;         // 12500

    // ---- iteration 0 ----
    class_kernel<<<CLS_GRID, 256, cls_lds, stream>>>(embedding_m, memberships,
                                                     h_cV, h_cW, upd, updh, acc);
    rel_kernel<<<rel_grid, 512, rel_lds, stream>>>(updh,
                                                   h_sVs, h_sVo, h_sWs, h_sWo,
                                                   h_oVs, h_oVo, h_oWs, h_oWo,
                                                   sub_w, obj_w, s_idx, o_idx, acc);
    fin_kernel<<<fin_grid, 256, 0, stream>>>(upd, acc, emb_ws);

    // ---- iteration 1 (writes final output) ----
    class_kernel<<<CLS_GRID, 256, cls_lds, stream>>>(emb_ws, memberships,
                                                     h_cV, h_cW, upd, updh, acc);
    rel_kernel<<<rel_grid, 512, rel_lds, stream>>>(updh,
                                                   h_sVs, h_sVo, h_sWs, h_sWo,
                                                   h_oVs, h_oVo, h_oWs, h_oWo,
                                                   sub_w, obj_w, s_idx, o_idx, acc);
    fin_kernel<<<fin_grid, 256, 0, stream>>>(upd, acc, out);
}
